// HybridQuanvolutionTransformerQuantum_65481071405231
// MI455X (gfx1250) — compile-verified, matrix operand order checked by reading, not run on hardware
//
#include <hip/hip_runtime.h>
#include <hip/hip_bf16.h>
#include <math.h>

typedef __attribute__((ext_vector_type(16))) _Float16 v16h;
typedef __attribute__((ext_vector_type(8)))  float    v8f;

#define BATCH  4096
#define NPATCH (BATCH * 196)
#define KDIM   784
#define KPAD   800
#define NBLK   4

// ---------------------------------------------------------------------------
// Kernel 0: convert token_W [8,784] f32 -> f16 padded [8,800] (zeros in tail)
// ---------------------------------------------------------------------------
__global__ void prep_w_kernel(const float* __restrict__ W, _Float16* __restrict__ Wh) {
    int t = blockIdx.x * blockDim.x + threadIdx.x;
    if (t >= 8 * KPAD) return;
    int n = t / KPAD, k = t - n * KPAD;
    Wh[t] = (k < KDIM) ? (_Float16)W[n * KDIM + k] : (_Float16)0.0f;
}

// ---------------------------------------------------------------------------
// Kernel 1: quanvolution. One thread = one 2x2 patch -> 4-qubit circuit.
// State (16 complex amps) fully register-resident; writes f16 features.
// ---------------------------------------------------------------------------
__global__ void quanv_kernel(const float* __restrict__ x,
                             const float* __restrict__ qrx,
                             _Float16* __restrict__ feat) {
    int tid = blockIdx.x * blockDim.x + threadIdx.x;
    if (tid >= NPATCH) return;
    int b = tid / 196, p = tid - b * 196;
    int r = p / 14, cc = p - r * 14;
    const float* img = x + b * 784;
    float tt[4];
    tt[0] = img[(2*r)*28   + 2*cc];
    tt[1] = img[(2*r)*28   + 2*cc + 1];
    tt[2] = img[(2*r+1)*28 + 2*cc];
    tt[3] = img[(2*r+1)*28 + 2*cc + 1];

    float cw[4], sw[4];
#pragma unroll
    for (int w = 0; w < 4; ++w) { cw[w] = cosf(0.5f*tt[w]); sw[w] = sinf(0.5f*tt[w]); }

    // RY product state (real): wire 0 = MSB of index
    float re[16], im[16];
#pragma unroll
    for (int i = 0; i < 16; ++i) {
        float a = 1.0f;
#pragma unroll
        for (int w = 0; w < 4; ++w) a *= ((i >> (3-w)) & 1) ? sw[w] : cw[w];
        re[i] = a; im[i] = 0.0f;
    }
    // RX on each wire: new = c*self + s*(i-rotated partner)
#pragma unroll
    for (int w = 0; w < 4; ++w) {
        float th = qrx[w];
        float c = cosf(0.5f*th), s = sinf(0.5f*th);
        int m = 1 << (3-w);
        float nr[16], ni[16];
#pragma unroll
        for (int i = 0; i < 16; ++i) {
            int j = i ^ m;
            nr[i] = c*re[i] + s*im[j];
            ni[i] = c*im[i] - s*re[j];
        }
#pragma unroll
        for (int i = 0; i < 16; ++i) { re[i] = nr[i]; im[i] = ni[i]; }
    }
    // CNOT ring (0,1),(1,2),(2,3),(3,0): pure register permutations
#pragma unroll
    for (int g = 0; g < 4; ++g) {
        int cm = 1 << (3 - g);
        int tm = 1 << (3 - ((g + 1) & 3));
#pragma unroll
        for (int i = 0; i < 16; ++i) {
            if ((i & cm) && !(i & tm)) {
                int j = i ^ tm;
                float t0 = re[i]; re[i] = re[j]; re[j] = t0;
                float t1 = im[i]; im[i] = im[j]; im[j] = t1;
            }
        }
    }
    float pr[16];
#pragma unroll
    for (int i = 0; i < 16; ++i) pr[i] = re[i]*re[i] + im[i]*im[i];

    _Float16* dst = feat + (size_t)b * KPAD + p * 4;
#pragma unroll
    for (int k = 0; k < 4; ++k) {
        float z = 0.0f;
#pragma unroll
        for (int i = 0; i < 16; ++i) z += ((i >> (3-k)) & 1) ? -pr[i] : pr[i];
        dst[k] = (_Float16)z;
    }
    // zero the K-pad region once per image (first 16 patch-threads)
    if (p < KPAD - KDIM) feat[(size_t)b * KPAD + KDIM + p] = (_Float16)0.0f;
}

// ---------------------------------------------------------------------------
// Kernel 2: token projection GEMM via v_wmma_f32_16x16x32_f16.
// One wave per 16-row tile; K padded to 800 -> 25 unguarded WMMA steps.
// Fragment map: elem e of lane l holds K = e + 8*(l>=16) + 8*(e>=8); M/N = l&15.
// ---------------------------------------------------------------------------
__global__ void token_gemm_kernel(const _Float16* __restrict__ feat,
                                  const _Float16* __restrict__ Wh,
                                  const float* __restrict__ bias,
                                  float* __restrict__ tok) {
    int wv    = (blockIdx.x * blockDim.x + threadIdx.x) >> 5;
    int lane  = threadIdx.x & 31;
    int mbase = wv * 16;
    if (mbase >= BATCH) return;
    int half = lane >> 4;
    int lmod = lane & 15;

    const _Float16* arow = feat + (size_t)(mbase + lmod) * KPAD;
    const _Float16* brow = Wh + (size_t)(lmod & 7) * KPAD;  // valid memory even if lmod>=8
    bool bvalid = (lmod < 8);

    v8f acc = {};
    for (int k0 = 0; k0 < KPAD; k0 += 32) {
        v16h a, bb;
#pragma unroll
        for (int e = 0; e < 16; ++e) {
            int k = k0 + e + 8*half + ((e >= 8) ? 8 : 0);
            a[e] = arow[k];
            _Float16 bv = brow[k];
            bb[e] = bvalid ? bv : (_Float16)0.0f;
        }
        acc = __builtin_amdgcn_wmma_f32_16x16x32_f16(false, a, false, bb,
                                                     (short)0, acc, false, false);
    }
    if (bvalid) {
        float bv = bias[lmod];
#pragma unroll
        for (int r = 0; r < 8; ++r) {
            int m = mbase + r + 8*half;
            tok[m * 8 + lmod] = acc[r] + bv;
        }
    }
}

// ---------------------------------------------------------------------------
// Kernel 3: transformer blocks. One wave32 per sample; the 8-qubit state
// (256 complex amps) lives as 8 complex values/lane. Lane bits = wires 0..4,
// slot bits = wires 5..7. Cross-lane gates via __shfl_xor.
// ---------------------------------------------------------------------------
__device__ __forceinline__ void layernorm8(const float* v, const float* g,
                                           const float* b, float* out) {
    float mu = 0.f;
#pragma unroll
    for (int k = 0; k < 8; ++k) mu += v[k];
    mu *= 0.125f;
    float var = 0.f;
#pragma unroll
    for (int k = 0; k < 8; ++k) { float d = v[k] - mu; var += d*d; }
    var *= 0.125f;
    float inv = 1.0f / sqrtf(var + 1e-5f);
#pragma unroll
    for (int k = 0; k < 8; ++k) out[k] = (v[k] - mu) * inv * g[k] + b[k];
}

__global__ void transformer_kernel(const float* __restrict__ tok,
    const float* __restrict__ attn_rx, const float* __restrict__ ffn_phi,
    const float* __restrict__ w1, const float* __restrict__ b1,
    const float* __restrict__ w2, const float* __restrict__ b2,
    const float* __restrict__ ln1g, const float* __restrict__ ln1b,
    const float* __restrict__ ln2g, const float* __restrict__ ln2b,
    const float* __restrict__ clsW, const float* __restrict__ clsb,
    float* __restrict__ out) {
    int wv   = (blockIdx.x * blockDim.x + threadIdx.x) >> 5;
    int lane = threadIdx.x & 31;
    if (wv >= BATCH) return;

    float h[8];
#pragma unroll
    for (int k = 0; k < 8; ++k)
        h[k] = tok[wv * 8 + k] + ((k & 1) ? 1.0f : 0.0f);   // + sinusoidal PE @ pos 0

    for (int blk = 0; blk < NBLK; ++blk) {
        // RX-encode h into an 8-qubit product state: amp(i) = (-i)^popc(i) * prod mags
        float cw[8], sw[8];
#pragma unroll
        for (int k = 0; k < 8; ++k) { cw[k] = cosf(0.5f*h[k]); sw[k] = sinf(0.5f*h[k]); }
        float re[8], im[8];
#pragma unroll
        for (int s = 0; s < 8; ++s) {
            int idx = lane * 8 + s;
            float mag = 1.0f;
#pragma unroll
            for (int w = 0; w < 8; ++w) mag *= ((idx >> (7-w)) & 1) ? sw[w] : cw[w];
            int pc = __popc(idx);
            float rs = (pc & 2) ? -mag : mag;
            float is = (pc & 2) ?  mag : -mag;
            re[s] = (pc & 1) ? 0.0f : rs;
            im[s] = (pc & 1) ? is : 0.0f;
        }
        // RX gates (commute): wires 0..4 cross-lane, wires 5..7 in-register
        const float* rx = attn_rx + blk * 8;
#pragma unroll
        for (int w = 0; w < 5; ++w) {
            float c = cosf(0.5f*rx[w]), s = sinf(0.5f*rx[w]);
            int mask = 1 << (4 - w);
#pragma unroll
            for (int q = 0; q < 8; ++q) {
                float pr = __shfl_xor(re[q], mask, 32);
                float pi = __shfl_xor(im[q], mask, 32);
                re[q] = c*re[q] + s*pi;
                im[q] = c*im[q] - s*pr;
            }
        }
#pragma unroll
        for (int w = 5; w < 8; ++w) {
            float c = cosf(0.5f*rx[w]), s = sinf(0.5f*rx[w]);
            int m = 1 << (7 - w);
            float nr[8], ni[8];
#pragma unroll
            for (int q = 0; q < 8; ++q) {
                int j = q ^ m;
                nr[q] = c*re[q] + s*im[j];
                ni[q] = c*im[q] - s*re[j];
            }
#pragma unroll
            for (int q = 0; q < 8; ++q) { re[q] = nr[q]; im[q] = ni[q]; }
        }
        // CNOT ring, in order: (0,1)(1,2)(2,3)(3,4) lane<->lane
#pragma unroll
        for (int g = 0; g < 4; ++g) {
            int cbit = 16 >> g, tmask = 8 >> g;
            bool ctl = (lane & cbit) != 0;
#pragma unroll
            for (int q = 0; q < 8; ++q) {
                float pr = __shfl_xor(re[q], tmask, 32);
                float pi = __shfl_xor(im[q], tmask, 32);
                re[q] = ctl ? pr : re[q];
                im[q] = ctl ? pi : im[q];
            }
        }
        { // (4,5): control = lane bit0, target = slot bit2
            bool ctl = (lane & 1) != 0;
#pragma unroll
            for (int q = 0; q < 4; ++q) {
                float tr = re[q], ti = im[q];
                re[q]   = ctl ? re[q+4] : re[q];
                im[q]   = ctl ? im[q+4] : im[q];
                re[q+4] = ctl ? tr : re[q+4];
                im[q+4] = ctl ? ti : im[q+4];
            }
        }
        { // (5,6): slots with bit2 -> swap (4,6),(5,7)
            float t;
            t=re[4]; re[4]=re[6]; re[6]=t;  t=im[4]; im[4]=im[6]; im[6]=t;
            t=re[5]; re[5]=re[7]; re[7]=t;  t=im[5]; im[5]=im[7]; im[7]=t;
        }
        { // (6,7): slots with bit1 -> swap (2,3),(6,7)
            float t;
            t=re[2]; re[2]=re[3]; re[3]=t;  t=im[2]; im[2]=im[3]; im[3]=t;
            t=re[6]; re[6]=re[7]; re[7]=t;  t=im[6]; im[6]=im[7]; im[7]=t;
        }
        // (7,0): odd slots exchange across lane mask 16
#pragma unroll
        for (int q = 1; q < 8; q += 2) {
            re[q] = __shfl_xor(re[q], 16, 32);
            im[q] = __shfl_xor(im[q], 16, 32);
        }
        // measure <Z_k>
        float p[8], tot = 0.f;
#pragma unroll
        for (int q = 0; q < 8; ++q) { p[q] = re[q]*re[q] + im[q]*im[q]; tot += p[q]; }
        float z[8];
#pragma unroll
        for (int k = 0; k < 5; ++k) z[k] = ((lane >> (4-k)) & 1) ? -tot : tot;
        z[5] = (p[0]+p[1]+p[2]+p[3]) - (p[4]+p[5]+p[6]+p[7]);
        z[6] = (p[0]+p[1]+p[4]+p[5]) - (p[2]+p[3]+p[6]+p[7]);
        z[7] = (p[0]+p[2]+p[4]+p[6]) - (p[1]+p[3]+p[5]+p[7]);
#pragma unroll
        for (int m = 1; m < 32; m <<= 1) {
#pragma unroll
            for (int k = 0; k < 8; ++k) z[k] += __shfl_xor(z[k], m, 32);
        }
        // residual + LN1
        float a1[8];
#pragma unroll
        for (int k = 0; k < 8; ++k) a1[k] = h[k] + z[k];
        layernorm8(a1, ln1g + blk*8, ln1b + blk*8, h);
        // quantum FFN: separable circuit -> cos(attn)*cos(phi), then 2 tiny GEMMs
        float qz[8];
#pragma unroll
        for (int k = 0; k < 8; ++k) qz[k] = cosf(z[k]) * cosf(ffn_phi[blk*8 + k]);
        float f[8];
#pragma unroll
        for (int k = 0; k < 8; ++k) f[k] = b2[blk*8 + k];
#pragma unroll
        for (int j = 0; j < 16; ++j) {
            float acc = b1[blk*16 + j];
#pragma unroll
            for (int k = 0; k < 8; ++k) acc += qz[k] * w1[blk*128 + j*8 + k];
            acc = fmaxf(acc, 0.0f);
#pragma unroll
            for (int k = 0; k < 8; ++k) f[k] += acc * w2[blk*128 + k*16 + j];
        }
#pragma unroll
        for (int k = 0; k < 8; ++k) a1[k] = h[k] + f[k];
        layernorm8(a1, ln2g + blk*8, ln2b + blk*8, h);
    }
    // classifier: lanes 0..9 each write one logit
    if (lane < 10) {
        float acc = clsb[lane];
#pragma unroll
        for (int k = 0; k < 8; ++k) acc += h[k] * clsW[lane*8 + k];
        out[wv * 10 + lane] = acc;
    }
}

// ---------------------------------------------------------------------------
extern "C" void kernel_launch(void* const* d_in, const int* in_sizes, int n_in,
                              void* d_out, int out_size, void* d_ws, size_t ws_size,
                              hipStream_t stream) {
    (void)in_sizes; (void)n_in; (void)out_size; (void)ws_size;
    const float* x        = (const float*)d_in[0];
    const float* quanv_rx = (const float*)d_in[1];
    const float* token_W  = (const float*)d_in[2];
    const float* token_b  = (const float*)d_in[3];
    const float* attn_rx  = (const float*)d_in[4];
    const float* ffn_phi  = (const float*)d_in[5];
    const float* ffn_w1   = (const float*)d_in[6];
    const float* ffn_b1   = (const float*)d_in[7];
    const float* ffn_w2   = (const float*)d_in[8];
    const float* ffn_b2   = (const float*)d_in[9];
    const float* ln1_g    = (const float*)d_in[10];
    const float* ln1_b    = (const float*)d_in[11];
    const float* ln2_g    = (const float*)d_in[12];
    const float* ln2_b    = (const float*)d_in[13];
    const float* cls_W    = (const float*)d_in[14];
    const float* cls_b    = (const float*)d_in[15];
    float* out = (float*)d_out;

    char* ws = (char*)d_ws;
    size_t featBytes = (size_t)BATCH * KPAD * sizeof(_Float16);   // 6.55 MB
    size_t whBytes   = (size_t)8 * KPAD * sizeof(_Float16);       // 12.8 KB
    _Float16* feat_h = (_Float16*)ws;
    _Float16* Wh     = (_Float16*)(ws + featBytes);
    float*    tok    = (float*)(ws + featBytes + whBytes);        // [4096][8]

    prep_w_kernel<<<(8 * KPAD + 255) / 256, 256, 0, stream>>>(token_W, Wh);
    quanv_kernel<<<(NPATCH + 255) / 256, 256, 0, stream>>>(x, quanv_rx, feat_h);
    token_gemm_kernel<<<(BATCH / 16 + 7) / 8, 256, 0, stream>>>(feat_h, Wh, token_b, tok);
    transformer_kernel<<<(BATCH + 7) / 8, 256, 0, stream>>>(tok, attn_rx, ffn_phi,
        ffn_w1, ffn_b1, ffn_w2, ffn_b2, ln1_g, ln1_b, ln2_g, ln2_b, cls_W, cls_b, out);
}
